// DGCNNBase_47897475285501
// MI455X (gfx1250) — compile-verified, matrix operand order checked by reading, not run on hardware
//
#include <hip/hip_runtime.h>

// ---------------------------------------------------------------------------
// DGCNN forward for MI455X (gfx1250, wave32, WMMA).
//  - KNN Gram matrix (the FLOP-dominant part) on V_WMMA_F32_16X16X4_F32
//    (full f32 precision -> neighbor ordering matches the f32 reference),
//    fused with a register-resident top-20 (no NxN matrix in memory).
//  - EdgeConv MLPs + final 192->1024 GEMM on V_WMMA_F32_16X16X32_F16.
//  - global_prefetch_b8 (via __builtin_prefetch) hides L2 latency on the
//    streamed candidate rows and the indexed neighbor gathers.
// ---------------------------------------------------------------------------

#define BB   2
#define NN   8192
#define KNB  20      // neighbors
#define EMB  1024

typedef _Float16 h16;
typedef __attribute__((ext_vector_type(16))) _Float16 v16h;
typedef __attribute__((ext_vector_type(8)))  _Float16 v8h;
typedef __attribute__((ext_vector_type(8)))  float    v8f;
typedef __attribute__((ext_vector_type(2)))  float    v2f;

// 16-bit A/B fragment of a 16x32 tile.  Lane L (g=L>>4, m=L&15) holds row m;
// element e holds K = kbase + 8*g + (e<8 ? e : e+8)  ->  two contiguous
// 16-byte chunks, so this lowers to 2x b128 loads (LDS or global).
__device__ __forceinline__ v16h load_frag(const h16* rowp, int g, int kbase) {
  const h16* p = rowp + kbase + 8 * g;
  v8h lo = *(const v8h*)(p);
  v8h hi = *(const v8h*)(p + 16);
  v16h a;
#pragma unroll
  for (int e = 0; e < 8; ++e) { a[e] = lo[e]; a[e + 8] = hi[e]; }
  return a;
}

__device__ __forceinline__ float lrelu(float x) { return x > 0.f ? x : 0.2f * x; }

// ---------------------------------------------------------------------------
// Prep: x0 = concat(xyz,xyz) [B,N,6] for edgeconv1, xyz4 = [B,N,4] zero-padded
// for f32-WMMA KNN frags, sqn = |xyz|^2 (6-dim dist = 2x 3-dim: same ordering).
__global__ void prep_all(const float* __restrict__ pts, float* __restrict__ x0,
                         float* __restrict__ xyz4, float* __restrict__ sqn) {
  int i = blockIdx.x * blockDim.x + threadIdx.x;
  if (i >= BB * NN) return;
  float x = pts[i * 3], y = pts[i * 3 + 1], z = pts[i * 3 + 2];
  x0[i * 6 + 0] = x; x0[i * 6 + 1] = y; x0[i * 6 + 2] = z;
  x0[i * 6 + 3] = x; x0[i * 6 + 4] = y; x0[i * 6 + 5] = z;
  xyz4[i * 4 + 0] = x; xyz4[i * 4 + 1] = y; xyz4[i * 4 + 2] = z; xyz4[i * 4 + 3] = 0.f;
  sqn[i] = x * x + y * y + z * z;
}

// f32 weights [Co,Ci] -> f16 padded to [Co,CiPad] (zero pad for K%32).
__global__ void convert_w(const float* __restrict__ w, h16* __restrict__ out,
                          int Co, int Ci, int CiPad) {
  int i = blockIdx.x * blockDim.x + threadIdx.x;
  if (i >= Co * CiPad) return;
  int o = i / CiPad, k = i % CiPad;
  out[i] = (k < Ci) ? (h16)w[o * Ci + k] : (h16)0.f;
}

__global__ void row_sqnorm64(const float* __restrict__ f, float* __restrict__ sqn) {
  int i = blockIdx.x * blockDim.x + threadIdx.x;
  if (i >= BB * NN) return;
  const float* r = f + (size_t)i * 64;
  float s = 0.f;
#pragma unroll
  for (int c = 0; c < 64; ++c) { float v = r[c]; s += v * v; }
  sqn[i] = s;
}

// ---------------------------------------------------------------------------
// WMMA KNN.  CP = padded feature stride (4 for xyz, 64 for x1/x2).
// Block = 128 threads (4 waves) owns 128 queries (8 M-tiles).  Query A-frags
// (f32 16x16x4: lane m=L&15 holds K = kc*4 + 2*(L>>4) + {0,1} -> one b64)
// are hoisted into registers for the whole kernel.  Candidates stream in
// 64-wide chunks (4 N-tiles); neg-distances land in a stride-65 LDS panel
// ((row+col)%64 banks -> conflict-free column scans); each thread then scans
// its query's row and bubble-inserts into a sorted register top-20.
// Next-chunk candidate rows are prefetched (global_prefetch_b8) while the
// current chunk's WMMA chain runs.  Strict '>' preserves JAX top_k
// earlier-index-first tie-break.
template <int CP>
__global__ __launch_bounds__(128) void knn_wmma(const float* __restrict__ feats,
                                                const float* __restrict__ sqn,
                                                int* __restrict__ idx) {
  const int QB = 128, CB = 64, KS = CP / 4, LDW = CB + 1;
  __shared__ float dist[QB * LDW];
  __shared__ float qs_s[QB];
  int b  = blockIdx.x / (NN / QB);
  int q0 = (blockIdx.x % (NN / QB)) * QB;
  const float* fb = feats + (size_t)b * NN * CP;
  const float* sb = sqn + (size_t)b * NN;
  int tid = threadIdx.x, wave = tid >> 5, lane = tid & 31;
  int g = (lane >> 4) & 1, lm = lane & 15;

  qs_s[tid] = sb[q0 + tid];
  __syncthreads();

  // Hoist this wave's query fragments (M-tiles 2*wave, 2*wave+1).
  v2f afr[2][KS];
#pragma unroll
  for (int m2 = 0; m2 < 2; ++m2) {
    int row = q0 + (wave * 2 + m2) * 16 + lm;
#pragma unroll
    for (int kc = 0; kc < KS; ++kc)
      afr[m2][kc] = *(const v2f*)(fb + (size_t)row * CP + kc * 4 + 2 * g);
  }

  float bd[KNB]; int bi[KNB];
#pragma unroll
  for (int t = 0; t < KNB; ++t) { bd[t] = -1e30f; bi[t] = 0; }

  for (int c0 = 0; c0 < NN; c0 += CB) {
    // Prefetch next chunk's candidate rows into cache while we compute.
    if (c0 + CB < NN) {
#pragma unroll
      for (int nt = 0; nt < 4; ++nt)
        __builtin_prefetch(fb + (size_t)(c0 + CB + nt * 16 + lm) * CP, 0, 3);
    }
#pragma unroll
    for (int nt = 0; nt < 4; ++nt) {
      int cn = c0 + nt * 16 + lm;
      v2f bfr[KS];
#pragma unroll
      for (int kc = 0; kc < KS; ++kc)
        bfr[kc] = *(const v2f*)(fb + (size_t)cn * CP + kc * 4 + 2 * g);
      float csn = sb[cn];
#pragma unroll
      for (int m2 = 0; m2 < 2; ++m2) {
        v8f acc = {0.f, 0.f, 0.f, 0.f, 0.f, 0.f, 0.f, 0.f};
#pragma unroll
        for (int kc = 0; kc < KS; ++kc)
          acc = __builtin_amdgcn_wmma_f32_16x16x4_f32(false, afr[m2][kc],
                                                      false, bfr[kc],
                                                      (short)0, acc, false, false);
#pragma unroll
        for (int r = 0; r < 8; ++r) {
          int rl = (wave * 2 + m2) * 16 + r + 8 * g;
          dist[rl * LDW + nt * 16 + lm] = 2.f * acc[r] - qs_s[rl] - csn;
        }
      }
    }
    __syncthreads();
    // Scan this thread's query row; maintain sorted top-20 in registers.
    for (int j = 0; j < CB; ++j) {
      float nd = dist[tid * LDW + j];
      if (nd > bd[KNB - 1]) {
        float d = nd; int id = c0 + j;
#pragma unroll
        for (int t = 0; t < KNB; ++t) {
          if (d > bd[t]) {
            float td = bd[t]; bd[t] = d; d = td;
            int   ti = bi[t]; bi[t] = id; id = ti;
          }
        }
      }
    }
    __syncthreads();
  }
  int* op = idx + ((size_t)b * NN + q0 + tid) * KNB;
#pragma unroll
  for (int t = 0; t < KNB; ++t) op[t] = bi[t];
}

// ---------------------------------------------------------------------------
// EdgeConv: per block P=4 points -> 80 rows of graph features (nb-ctr, ctr),
// layer a (K=CKA) and optional layer b (K=64) via WMMA, leaky-relu epilogs,
// then max over the 20 neighbor rows per point.  Neighbor rows are prefetched
// as soon as their indices are known (before the barrier) to warm the cache
// for the indexed gather.
template <int CIN, int CKA, bool TWO>
__global__ __launch_bounds__(256) void edgeconv_kernel(
    const float* __restrict__ feats, const int* __restrict__ idx,
    const h16* __restrict__ wa, const float* __restrict__ sa, const float* __restrict__ ba,
    const h16* __restrict__ wb, const float* __restrict__ sb, const float* __restrict__ bb,
    float* __restrict__ out) {
  const int P = 4, ROWS = P * KNB;           // 80 rows = 5 M-tiles
  __shared__ h16   gA[ROWS * CKA];           // graph features (f16)
  __shared__ h16   mid[ROWS * 64];           // layer-a output (f16)
  __shared__ float pan[ROWS * 64];           // final panel for max-over-k
  __shared__ int   sIdx[ROWS];
  __shared__ float sCtr[P * CIN];

  int b  = blockIdx.x / (NN / P);
  int n0 = (blockIdx.x % (NN / P)) * P;
  const float* fb = feats + (size_t)b * NN * CIN;
  int tid = threadIdx.x;

  if (tid < ROWS) {
    int nb = idx[((size_t)b * NN + n0 + tid / KNB) * KNB + tid % KNB];
    sIdx[tid] = nb;
    __builtin_prefetch(fb + (size_t)nb * CIN, 0, 3);   // warm gather rows
  }
  for (int e = tid; e < P * CIN; e += 256)
    sCtr[e] = fb[(size_t)(n0 + e / CIN) * CIN + e % CIN];
  __syncthreads();

  for (int e = tid; e < ROWS * CIN; e += 256) {
    int row = e / CIN, c = e % CIN, p = row / KNB;
    float ctr = sCtr[p * CIN + c];
    float nb  = fb[(size_t)sIdx[row] * CIN + c];
    gA[row * CKA + c]       = (h16)(nb - ctr);
    gA[row * CKA + CIN + c] = (h16)ctr;
  }
  if (2 * CIN < CKA) {
    const int PADC = CKA - 2 * CIN;
    for (int e = tid; e < ROWS * PADC; e += 256)
      gA[(e / PADC) * CKA + 2 * CIN + (e % PADC)] = (h16)0.f;
  }
  __syncthreads();

  int wave = tid >> 5, lane = tid & 31, g = (lane >> 4) & 1, lm = lane & 15;
  const int MT = ROWS / 16, NT = 4, TILES = MT * NT;   // 20 tiles, 8 waves

  // Layer a: [80 x CKA] x [CKA x 64]
  for (int t = wave; t < TILES; t += 8) {
    int mt = t / NT, nt = t % NT;
    v8f acc = {0.f, 0.f, 0.f, 0.f, 0.f, 0.f, 0.f, 0.f};
    const h16* arow = gA + (mt * 16 + lm) * CKA;
    const h16* brow = wa + (size_t)(nt * 16 + lm) * CKA;
#pragma unroll
    for (int kb = 0; kb < CKA; kb += 32) {
      v16h af = load_frag(arow, g, kb);
      v16h bf = load_frag(brow, g, kb);
      acc = __builtin_amdgcn_wmma_f32_16x16x32_f16(false, af, false, bf,
                                                   (short)0, acc, false, false);
    }
    int ch = nt * 16 + lm;
    float s = sa[ch], bias = ba[ch];
#pragma unroll
    for (int r = 0; r < 8; ++r) {
      int row = mt * 16 + r + 8 * g;
      float v = lrelu(acc[r] * s + bias);
      if (TWO) mid[row * 64 + ch] = (h16)v; else pan[row * 64 + ch] = v;
    }
  }
  __syncthreads();

  if (TWO) {  // Layer b: [80 x 64] x [64 x 64]
    for (int t = wave; t < TILES; t += 8) {
      int mt = t / NT, nt = t % NT;
      v8f acc = {0.f, 0.f, 0.f, 0.f, 0.f, 0.f, 0.f, 0.f};
      const h16* arow = mid + (mt * 16 + lm) * 64;
      const h16* brow = wb + (size_t)(nt * 16 + lm) * 64;
#pragma unroll
      for (int kb = 0; kb < 64; kb += 32) {
        v16h af = load_frag(arow, g, kb);
        v16h bf = load_frag(brow, g, kb);
        acc = __builtin_amdgcn_wmma_f32_16x16x32_f16(false, af, false, bf,
                                                     (short)0, acc, false, false);
      }
      int ch = nt * 16 + lm;
      float s = sb[ch], bias = bb[ch];
#pragma unroll
      for (int r = 0; r < 8; ++r) {
        int row = mt * 16 + r + 8 * g;
        pan[row * 64 + ch] = lrelu(acc[r] * s + bias);
      }
    }
    __syncthreads();
  }

  // Max over the 20 neighbor rows: 256 threads = 4 points x 64 channels.
  {
    int p = tid >> 6, ch = tid & 63;
    float m = -1e30f;
#pragma unroll
    for (int kk = 0; kk < KNB; ++kk) m = fmaxf(m, pan[(p * KNB + kk) * 64 + ch]);
    out[((size_t)b * NN + n0 + p) * 64 + ch] = m;
  }
}

// ---------------------------------------------------------------------------
// Final 192 -> 1024 GEMM with fused per-channel partial max (over the block's
// 32 points).  Channels are partitioned across waves -> no reduction races.
__global__ __launch_bounds__(256) void final_partial(
    const float* __restrict__ x1, const float* __restrict__ x2,
    const float* __restrict__ x3, const h16* __restrict__ wg,
    const float* __restrict__ sg, const float* __restrict__ bg,
    float* __restrict__ partial) {  // [B][N/32][1024]
  const int P = 32;
  __shared__ h16 catA[P * 192];
  int b = blockIdx.x / (NN / P);
  int blk = blockIdx.x % (NN / P);
  int n0 = blk * P;
  int tid = threadIdx.x;
  for (int e = tid; e < P * 192; e += 256) {
    int row = e / 192, c = e % 192;
    const float* src = (c < 64) ? x1 : (c < 128) ? x2 : x3;
    catA[e] = (h16)src[((size_t)b * NN + n0 + row) * 64 + (c & 63)];
  }
  __syncthreads();
  int wave = tid >> 5, lane = tid & 31, g = (lane >> 4) & 1, lm = lane & 15;
  for (int nt = wave; nt < 64; nt += 8) {
    int ch = nt * 16 + lm;
    float s = sg[ch], bias = bg[ch];
    float cmax = -1e30f;
#pragma unroll
    for (int mt = 0; mt < 2; ++mt) {
      v8f acc = {0.f, 0.f, 0.f, 0.f, 0.f, 0.f, 0.f, 0.f};
      const h16* arow = catA + (mt * 16 + lm) * 192;
      const h16* brow = wg + (size_t)ch * 192;
#pragma unroll
      for (int kb = 0; kb < 192; kb += 32) {
        v16h af = load_frag(arow, g, kb);
        v16h bf = load_frag(brow, g, kb);
        acc = __builtin_amdgcn_wmma_f32_16x16x32_f16(false, af, false, bf,
                                                     (short)0, acc, false, false);
      }
#pragma unroll
      for (int r = 0; r < 8; ++r) cmax = fmaxf(cmax, lrelu(acc[r] * s + bias));
    }
    cmax = fmaxf(cmax, __shfl_xor(cmax, 16));  // merge the two row-groups
    if (lane < 16)
      partial[((size_t)(b * (NN / P) + blk)) * EMB + ch] = cmax;
  }
}

__global__ void reduce_partials(const float* __restrict__ partial,
                                float* __restrict__ gmax) {
  int b = blockIdx.x, c = threadIdx.x;   // blockDim = 1024
  float m = -1e30f;
  for (int j = 0; j < NN / 32; ++j)
    m = fmaxf(m, partial[((size_t)(b * (NN / 32) + j)) * EMB + c]);
  gmax[b * EMB + c] = m;
}

__global__ void broadcast_out(const float* __restrict__ gmax,
                              float* __restrict__ out) {
  size_t i = (size_t)blockIdx.x * blockDim.x + threadIdx.x;  // one float4 each
  size_t total = (size_t)BB * EMB * NN / 4;
  if (i >= total) return;
  size_t bc = i / (NN / 4);
  float v = gmax[bc];
  float4 vv = {v, v, v, v};
  ((float4*)out)[i] = vv;
}

// ---------------------------------------------------------------------------
static inline char* carve(char*& p, size_t bytes) {
  char* r = p;
  p += (bytes + 255) & ~(size_t)255;
  return r;
}

extern "C" void kernel_launch(void* const* d_in, const int* in_sizes, int n_in,
                              void* d_out, int out_size, void* d_ws, size_t ws_size,
                              hipStream_t stream) {
  (void)in_sizes; (void)n_in; (void)out_size; (void)ws_size;
  const float* pts = (const float*)d_in[0];
  const float* w1a = (const float*)d_in[1];  const float* s1a = (const float*)d_in[2];  const float* b1a = (const float*)d_in[3];
  const float* w1b = (const float*)d_in[4];  const float* s1b = (const float*)d_in[5];  const float* b1b = (const float*)d_in[6];
  const float* w2a = (const float*)d_in[7];  const float* s2a = (const float*)d_in[8];  const float* b2a = (const float*)d_in[9];
  const float* w2b = (const float*)d_in[10]; const float* s2b = (const float*)d_in[11]; const float* b2b = (const float*)d_in[12];
  const float* w3a = (const float*)d_in[13]; const float* s3a = (const float*)d_in[14]; const float* b3a = (const float*)d_in[15];
  const float* wg  = (const float*)d_in[16]; const float* sg  = (const float*)d_in[17]; const float* bg  = (const float*)d_in[18];

  char* p = (char*)d_ws;
  float* x0   = (float*)carve(p, (size_t)BB * NN * 6 * 4);
  float* xyz4 = (float*)carve(p, (size_t)BB * NN * 4 * 4);
  float* sqn  = (float*)carve(p, (size_t)BB * NN * 4);
  int*   idx  = (int*)  carve(p, (size_t)BB * NN * KNB * 4);
  float* x1   = (float*)carve(p, (size_t)BB * NN * 64 * 4);
  float* x2   = (float*)carve(p, (size_t)BB * NN * 64 * 4);
  float* x3   = (float*)carve(p, (size_t)BB * NN * 64 * 4);
  h16* w1a_h  = (h16*)  carve(p, (size_t)64 * 32 * 2);
  h16* w1b_h  = (h16*)  carve(p, (size_t)64 * 64 * 2);
  h16* w2a_h  = (h16*)  carve(p, (size_t)64 * 128 * 2);
  h16* w2b_h  = (h16*)  carve(p, (size_t)64 * 64 * 2);
  h16* w3a_h  = (h16*)  carve(p, (size_t)64 * 128 * 2);
  h16* wg_h   = (h16*)  carve(p, (size_t)EMB * 192 * 2);
  float* part = (float*)carve(p, (size_t)BB * (NN / 32) * EMB * 4);
  float* gmax = (float*)carve(p, (size_t)BB * EMB * 4);

  prep_all<<<(BB * NN + 255) / 256, 256, 0, stream>>>(pts, x0, xyz4, sqn);
  convert_w<<<(64 * 32 + 255) / 256, 256, 0, stream>>>(w1a, w1a_h, 64, 12, 32);
  convert_w<<<(64 * 64 + 255) / 256, 256, 0, stream>>>(w1b, w1b_h, 64, 64, 64);
  convert_w<<<(64 * 128 + 255) / 256, 256, 0, stream>>>(w2a, w2a_h, 64, 128, 128);
  convert_w<<<(64 * 64 + 255) / 256, 256, 0, stream>>>(w2b, w2b_h, 64, 64, 64);
  convert_w<<<(64 * 128 + 255) / 256, 256, 0, stream>>>(w3a, w3a_h, 64, 128, 128);
  convert_w<<<(EMB * 192 + 255) / 256, 256, 0, stream>>>(wg, wg_h, EMB, 192, 192);

  knn_wmma<4><<<BB * NN / 128, 128, 0, stream>>>(xyz4, sqn, idx);
  edgeconv_kernel<6, 32, true><<<BB * NN / 4, 256, 0, stream>>>(
      x0, idx, w1a_h, s1a, b1a, w1b_h, s1b, b1b, x1);

  row_sqnorm64<<<(BB * NN + 255) / 256, 256, 0, stream>>>(x1, sqn);
  knn_wmma<64><<<BB * NN / 128, 128, 0, stream>>>(x1, sqn, idx);
  edgeconv_kernel<64, 128, true><<<BB * NN / 4, 256, 0, stream>>>(
      x1, idx, w2a_h, s2a, b2a, w2b_h, s2b, b2b, x2);

  row_sqnorm64<<<(BB * NN + 255) / 256, 256, 0, stream>>>(x2, sqn);
  knn_wmma<64><<<BB * NN / 128, 128, 0, stream>>>(x2, sqn, idx);
  edgeconv_kernel<64, 128, false><<<BB * NN / 4, 256, 0, stream>>>(
      x2, idx, w3a_h, s3a, b3a, nullptr, nullptr, nullptr, x3);

  final_partial<<<BB * NN / 32, 256, 0, stream>>>(x1, x2, x3, wg_h, sg, bg, part);
  reduce_partials<<<BB, 1024, 0, stream>>>(part, gmax);
  size_t tot4 = (size_t)BB * EMB * NN / 4;
  broadcast_out<<<(tot4 + 255) / 256, 256, 0, stream>>>(gmax, (float*)d_out);
}
